// AttentionEncoder_32830730010913
// MI455X (gfx1250) — matrix operand order repaired and checked by reading, not compiled, not run on hardware
//
#include <hip/hip_runtime.h>

typedef __attribute__((ext_vector_type(16))) _Float16 v16h;
typedef __attribute__((ext_vector_type(8)))  _Float16 v8h;
typedef __attribute__((ext_vector_type(4)))  _Float16 v4h;
typedef __attribute__((ext_vector_type(8)))  float    v8f;
typedef __attribute__((ext_vector_type(4)))  float    v4f;

#define WAVES_PER_BLOCK 4
#define DIM  128
#define KNBR 32
#define ROWSTRIDE (DIM + 8)   // 136 halves = 272 B row pitch (bank-conflict padding)

__global__ __launch_bounds__(32 * WAVES_PER_BLOCK)
void attn_encoder_kernel(const float* __restrict__ x,
                         const float* __restrict__ nb,
                         float* __restrict__ out,
                         int tokens)
{
    __shared__ _Float16 tile_s[WAVES_PER_BLOCK][KNBR * ROWSTRIDE];
    __shared__ _Float16 xh_s[WAVES_PER_BLOCK][DIM];
    __shared__ float    attn_s[WAVES_PER_BLOCK][KNBR];

    const int lane = threadIdx.x & 31;
    const int wv   = threadIdx.x >> 5;
    const int token = blockIdx.x * WAVES_PER_BLOCK + wv;
    if (token >= tokens) return;            // wave-uniform: EXEC stays all-ones

    const float* xp = x  + (size_t)token * DIM;
    const float* np = nb + (size_t)token * KNBR * DIM;
    float*       op = out + (size_t)token * DIM;

    _Float16* tw = tile_s[wv];
    _Float16* xw = xh_s[wv];

    // ---- x: keep f32 residual in regs, f16 copy to LDS (lane owns d = 4*lane..) ----
    v4f xv = *(const v4f*)(xp + lane * 4);
    {
        v4h xhv;
        #pragma unroll
        for (int i = 0; i < 4; ++i) xhv[i] = (_Float16)xv[i];
        *(v4h*)(xw + lane * 4) = xhv;
    }

    // ---- neighbor tile: coalesced f32 load -> f16 row-major padded LDS tile ----
    #pragma unroll 8
    for (int r = 0; r < KNBR; ++r) {
        v4f t = *(const v4f*)(np + r * DIM + lane * 4);
        v4h th;
        #pragma unroll
        for (int i = 0; i < 4; ++i) th[i] = (_Float16)t[i];
        *(v4h*)(tw + r * ROWSTRIDE + lane * 4) = th;
    }
    asm volatile("s_wait_dscnt 0" ::: "memory");   // our stores visible before cross-lane reads

    const int half = lane >> 4;     // 0: K-elems {0..15} for both A and B (contiguous)
    const int lj   = lane & 15;

    // ---- Stage 1: scores via v_wmma_f32_16x16x32_f16 ----
    // D[m,n] = sum_d A[m,d]*B[d,n];  A = x replicated over m, B = neighbor rows (col n = row n)
    v8f c0 = {};   // n = k 0..15
    v8f c1 = {};   // n = k 16..31
    #pragma unroll
    for (int kk = 0; kk < 4; ++kk) {
        const int d0 = kk * 32;
        // A (16-bit 16x32 layout): half0 elems = x[d0+0..15]; half1 = x[d0+16..31] (contig, matches B)
        v8h alo = *(const v8h*)(xw + d0 + half * 16);
        v8h ahi = *(const v8h*)(xw + d0 + half * 16 + 8);
        v16h a = __builtin_shufflevector(alo, ahi,
                  0,1,2,3,4,5,6,7, 8,9,10,11,12,13,14,15);
        // B (16-bit 32x16 layout): lane col n=lj, half0 K=d0+0..15, half1 K=d0+16..31 (contig rows)
        const _Float16* b0p = tw + lj * ROWSTRIDE + d0 + half * 16;
        const _Float16* b1p = tw + (16 + lj) * ROWSTRIDE + d0 + half * 16;
        v8h b0lo = *(const v8h*)(b0p);
        v8h b0hi = *(const v8h*)(b0p + 8);
        v8h b1lo = *(const v8h*)(b1p);
        v8h b1hi = *(const v8h*)(b1p + 8);
        v16h b0 = __builtin_shufflevector(b0lo, b0hi,
                  0,1,2,3,4,5,6,7, 8,9,10,11,12,13,14,15);
        v16h b1 = __builtin_shufflevector(b1lo, b1hi,
                  0,1,2,3,4,5,6,7, 8,9,10,11,12,13,14,15);
        c0 = __builtin_amdgcn_wmma_f32_16x16x32_f16(false, a, false, b0,
                                                    (short)0, c0, false, false);
        c1 = __builtin_amdgcn_wmma_f32_16x16x32_f16(false, a, false, b1,
                                                    (short)0, c1, false, false);
    }

    // ---- Softmax over 32 scores (lane j holds s[j%16] and s[j%16+16]; halves duplicate) ----
    const float scale = 0.088388347648318447f;     // 1/sqrt(128)
    float s0 = c0[0] * scale;
    float s1 = c1[0] * scale;
    float mx = fmaxf(s0, s1);
    #pragma unroll
    for (int off = 8; off >= 1; off >>= 1)
        mx = fmaxf(mx, __shfl_xor(mx, off, 32));
    float e0 = __expf(s0 - mx);
    float e1 = __expf(s1 - mx);
    float sum = e0 + e1;
    #pragma unroll
    for (int off = 8; off >= 1; off >>= 1)
        sum += __shfl_xor(sum, off, 32);
    float inv = 1.0f / sum;
    float a0 = e0 * inv;
    float a1 = e1 * inv;

    if (half == 0) {                     // one copy of attn weights to LDS
        attn_s[wv][lj]      = a0;
        attn_s[wv][16 + lj] = a1;
    }
    asm volatile("s_wait_dscnt 0" ::: "memory");

    // ---- Stage 2: out = x + sum_k attn[k] * neighbor[k,:]  (f32 accumulate) ----
    v4f acc = xv;                        // residual
    #pragma unroll 8
    for (int k = 0; k < KNBR; ++k) {
        float w = attn_s[wv][k];         // broadcast LDS read
        v4h nh = *(const v4h*)(tw + k * ROWSTRIDE + lane * 4);  // conflict-free b64
        acc[0] += w * (float)nh[0];
        acc[1] += w * (float)nh[1];
        acc[2] += w * (float)nh[2];
        acc[3] += w * (float)nh[3];
    }
    *(v4f*)(op + lane * 4) = acc;        // coalesced b128 store
}

extern "C" void kernel_launch(void* const* d_in, const int* in_sizes, int n_in,
                              void* d_out, int out_size, void* d_ws, size_t ws_size,
                              hipStream_t stream) {
    const float* x  = (const float*)d_in[0];   // [B,N,D] f32
    const float* nb = (const float*)d_in[1];   // [B,N,K,D] f32
    float* out = (float*)d_out;                // [B,N,D] f32
    const int tokens = in_sizes[0] / DIM;      // B*N = 80000
    dim3 grid((tokens + WAVES_PER_BLOCK - 1) / WAVES_PER_BLOCK);
    dim3 block(32 * WAVES_PER_BLOCK);
    hipLaunchKernelGGL(attn_encoder_kernel, grid, block, 0, stream, x, nb, out, tokens);
}
